// CausalSelfAttention_73821897883867
// MI455X (gfx1250) — matrix operand order repaired and checked by reading, not compiled, not run on hardware
//
#include <hip/hip_runtime.h>
#include <cstdint>
#include <cstddef>

#define B_   2
#define S_   2048
#define DM_  2048
#define HQ_  16
#define HKV_ 8
#define HD_  128

typedef __bf16 bf16;
typedef bf16  v16bf __attribute__((ext_vector_type(16)));
typedef float v8f   __attribute__((ext_vector_type(8)));

union Frag { v16bf v; unsigned u[8]; };

__device__ __forceinline__ v8f wmma_bf16(v16bf a, v16bf b, v8f c) {
  // D = A(16x32 bf16) * B(32x16 bf16) + C(16x16 f32)
  return __builtin_amdgcn_wmma_f32_16x16x32_bf16(false, a, false, b, (short)0, c,
                                                 false, false);
}

__device__ __forceinline__ v8f vzero8() {
  v8f x;
#pragma unroll
  for (int i = 0; i < 8; ++i) x[i] = 0.0f;
  return x;
}

// CDNA5 async copy: global -> LDS, 16B per lane, tracked on ASYNCcnt.
__device__ __forceinline__ void async_b128(const void* sbase, unsigned voff,
                                           unsigned ldsoff) {
  asm volatile("global_load_async_to_lds_b128 %0, %1, %2 offset:0"
               :: "v"(ldsoff), "v"(voff), "s"(sbase) : "memory");
}
__device__ __forceinline__ void wait_async0() {
  asm volatile("s_wait_asynccnt 0" ::: "memory");
}

// ---------------------------------------------------------------- conversions
__global__ void cvt_f32_bf16(const float* __restrict__ in, bf16* __restrict__ out,
                             long n) {
  long i = (long)blockIdx.x * blockDim.x + threadIdx.x;
  long stride = (long)gridDim.x * blockDim.x;
  for (; i < n; i += stride) out[i] = (bf16)in[i];
}

// ---------------------------------------------------------------- RoPE (bf16)
// Tensor layout (B, S, H, HD). One thread per (b,s,h, pair).
__global__ void rope_bf16(bf16* __restrict__ t, int H, long npairs) {
  long idx = (long)blockIdx.x * blockDim.x + threadIdx.x;
  if (idx >= npairs) return;
  int i = (int)(idx & 63);          // pair index within head (HD/2 = 64)
  long rem = idx >> 6;              // (b*S + s)*H + h
  long bs = rem / H;
  int s = (int)(bs % S_);
  // freq = theta^(-2i/HD); ln(10000)=9.210340372
  float freq = __expf((float)i * (-2.0f * 9.210340372f / (float)HD_));
  float ang = (float)s * freq;
  float sn, cs;
  __sincosf(ang, &sn, &cs);
  bf16* p = t + idx * 2;
  float xr = (float)p[0], xi = (float)p[1];
  p[0] = (bf16)(xr * cs - xi * sn);
  p[1] = (bf16)(xr * sn + xi * cs);
}

// ------------------------------------------------- NT GEMM: C = A(MxK)*W(NxK)^T
// Block: 256 threads = 8 waves. Block tile 128(M) x 128(N), K-step 32.
// Wave (wm 0..3, wn 0..1) computes 32x64 = 2x4 tiles of 16x16.
// Tiles staged with global_load_async_to_lds_b128 (ASYNCcnt path).
template <bool OUT_BF16>
__global__ __launch_bounds__(256)
void gemm_nt_bf16(const bf16* __restrict__ A, const bf16* __restrict__ W,
                  void* __restrict__ C, int M, int N, int K) {
  __shared__ bf16 sA[128][32];
  __shared__ bf16 sB[128][32];
  const int tid  = threadIdx.x;
  const int lane = tid & 31;
  const int wave = tid >> 5;
  const int wm = wave >> 1;          // 0..3
  const int wn = wave & 1;           // 0..1
  const int bm = blockIdx.y * 128;
  const int bn = blockIdx.x * 128;
  const int nlane = lane & 15;
  const int gA   = (lane & 16) ? 8 : 0;   // A K-group offset
  const int gB   = (lane & 16) ? 16 : 0;  // B K offset
  const int hi8  = (lane & 16) ? 8 : 0;   // C row offset

  v8f acc[2][4];
#pragma unroll
  for (int mi = 0; mi < 2; ++mi)
#pragma unroll
    for (int ni = 0; ni < 4; ++ni) acc[mi][ni] = vzero8();

  for (int k0 = 0; k0 < K; k0 += 32) {
    // async-stage 128x32 bf16 tiles (4x 16B chunks per row)
#pragma unroll
    for (int i = tid; i < 128 * 4; i += 256) {
      int r = i >> 2, c = i & 3;
      unsigned ga = (unsigned)(((size_t)(bm + r) * K + k0 + c * 8) * 2);
      unsigned gb = (unsigned)(((size_t)(bn + r) * K + k0 + c * 8) * 2);
      async_b128(A, ga, (unsigned)(uintptr_t)(&sA[r][0] + c * 8));
      async_b128(W, gb, (unsigned)(uintptr_t)(&sB[r][0] + c * 8));
    }
    wait_async0();
    __syncthreads();

    const unsigned* sA32 = (const unsigned*)sA;  // 16 dwords per row
    const unsigned* sB32 = (const unsigned*)sB;
    Frag a[2], b[4];
#pragma unroll
    for (int mi = 0; mi < 2; ++mi) {
      int base = (wm * 32 + mi * 16 + nlane) * 16;
#pragma unroll
      for (int j = 0; j < 8; ++j) {
        int kd = gA + j;
        a[mi].u[j] = sA32[base + kd];
      }
    }
#pragma unroll
    for (int ni = 0; ni < 4; ++ni) {
      int base = (wn * 64 + ni * 16 + nlane) * 16;
#pragma unroll
      for (int j = 0; j < 8; ++j) b[ni].u[j] = sB32[base + (gB >> 1) + j];
    }
#pragma unroll
    for (int mi = 0; mi < 2; ++mi)
#pragma unroll
      for (int ni = 0; ni < 4; ++ni)
        acc[mi][ni] = wmma_bf16(a[mi].v, b[ni].v, acc[mi][ni]);
    __syncthreads();
  }

#pragma unroll
  for (int mi = 0; mi < 2; ++mi)
#pragma unroll
    for (int ni = 0; ni < 4; ++ni)
#pragma unroll
      for (int r = 0; r < 8; ++r) {
        int row = bm + wm * 32 + mi * 16 + r + hi8;
        int col = bn + wn * 64 + ni * 16 + nlane;
        if (OUT_BF16)
          ((bf16*)C)[(size_t)row * N + col] = (bf16)acc[mi][ni][r];
        else
          ((float*)C)[(size_t)row * N + col] = acc[mi][ni][r];
      }
}

// ------------------------------------------------------------- flash attention
// Grid: (S/64, HQ, B). Block: 128 threads = 4 waves; wave w owns 16 query rows.
// K staged async (row-major); V staged transposed so P*V B-fragments are
// contiguous packed-K dwords (merge into ds_load_b128).
__global__ __launch_bounds__(128)
void flash_attn_bf16(const bf16* __restrict__ q, const bf16* __restrict__ k,
                     const bf16* __restrict__ v, bf16* __restrict__ o) {
  __shared__ bf16 sK[64][HD_];      // 16 KB, [key][dim]
  __shared__ bf16 sVt[HD_][64];     // 16 KB, [dim][key]
  __shared__ bf16 sP[4][16][64];    // 8 KB, per-wave region

  const int tid  = threadIdx.x;
  const int lane = tid & 31;
  const int wid  = tid >> 5;
  const int qb   = blockIdx.x * 64;
  const int h    = blockIdx.y;
  const int b    = blockIdx.z;
  const int hk   = h >> 1;                 // GQA: HQ/HKV = 2
  const int nlane = lane & 15;
  const int hi8   = (lane & 16) ? 8 : 0;
  const int gA    = (lane & 16) ? 8 : 0;
  const int gB    = (lane & 16) ? 16 : 0;
  const float scale = 0.08838834764831845f;  // 1/sqrt(128)

  // Q fragments for 4 K-steps of 32 (head dim 128), direct from global.
  Frag aq[4];
  {
    int qrow = qb + wid * 16 + nlane;
    const unsigned* qg =
        (const unsigned*)(q + (((size_t)b * S_ + qrow) * HQ_ + h) * HD_);
#pragma unroll
    for (int ks = 0; ks < 4; ++ks)
#pragma unroll
      for (int j = 0; j < 8; ++j) {
        int kd = ks * 16 + gA + j;
        aq[ks].u[j] = qg[kd];
      }
  }

  v8f oacc[8];
#pragma unroll
  for (int ni = 0; ni < 8; ++ni) oacc[ni] = vzero8();
  float m[8], l[8];
#pragma unroll
  for (int r = 0; r < 8; ++r) { m[r] = -1e30f; l[r] = 0.0f; }

  const int tmax = blockIdx.x;  // inclusive (causal)
  for (int t = 0; t <= tmax; ++t) {
    // K: async 16B chunks; V: load + transpose into sVt
#pragma unroll
    for (int i = tid; i < 64 * 16; i += 128) {
      int r = i >> 4, c = i & 15;
      size_t eoff = (((size_t)b * S_ + (t * 64 + r)) * HKV_ + hk) * HD_ + c * 8;
      async_b128(k, (unsigned)(eoff * 2), (unsigned)(uintptr_t)&sK[r][c * 8]);
      uint4 d = *(const uint4*)(v + eoff);
      const bf16* e = (const bf16*)&d;
#pragma unroll
      for (int j = 0; j < 8; ++j) sVt[c * 8 + j][r] = e[j];
    }
    wait_async0();
    __syncthreads();

    // S = Q * K^T  (4 N-tiles of 16 keys, K-dim = head dim 128)
    const unsigned* sK32 = (const unsigned*)sK;  // 64 dwords per key row
    v8f sacc[4];
#pragma unroll
    for (int ni = 0; ni < 4; ++ni) sacc[ni] = vzero8();
#pragma unroll
    for (int ks = 0; ks < 4; ++ks)
#pragma unroll
      for (int ni = 0; ni < 4; ++ni) {
        Frag bk;
        int krow = ni * 16 + nlane;
#pragma unroll
        for (int j = 0; j < 8; ++j)
          bk.u[j] = sK32[krow * 64 + ks * 16 + (gB >> 1) + j];
        sacc[ni] = wmma_bf16(aq[ks].v, bk.v, sacc[ni]);
      }

    // scale + causal mask + online softmax
    const bool diag = (t == tmax);
    float mnew[8], alpha[8], lsum[8];
#pragma unroll
    for (int r = 0; r < 8; ++r) {
      int qg_ = qb + wid * 16 + r + hi8;
      float mx = -1e30f;
#pragma unroll
      for (int ni = 0; ni < 4; ++ni) {
        float s = sacc[ni][r] * scale;
        if (diag && (t * 64 + ni * 16 + nlane) > qg_) s = -1e30f;
        sacc[ni][r] = s;
        mx = fmaxf(mx, s);
      }
#pragma unroll
      for (int off = 1; off < 16; off <<= 1)
        mx = fmaxf(mx, __shfl_xor(mx, off, 32));
      mnew[r] = fmaxf(m[r], mx);
      alpha[r] = __expf(m[r] - mnew[r]);
      m[r] = mnew[r];
      lsum[r] = 0.0f;
    }
#pragma unroll
    for (int ni = 0; ni < 4; ++ni)
#pragma unroll
      for (int r = 0; r < 8; ++r) {
        float p = __expf(sacc[ni][r] - m[r]);
        sacc[ni][r] = p;
        lsum[r] += p;
      }
#pragma unroll
    for (int r = 0; r < 8; ++r) {
#pragma unroll
      for (int off = 1; off < 16; off <<= 1)
        lsum[r] += __shfl_xor(lsum[r], off, 32);
      l[r] = l[r] * alpha[r] + lsum[r];
    }
#pragma unroll
    for (int ni = 0; ni < 8; ++ni)
#pragma unroll
      for (int r = 0; r < 8; ++r) oacc[ni][r] *= alpha[r];

    // P: C-layout regs -> LDS -> A-layout fragments (per-wave region)
#pragma unroll
    for (int ni = 0; ni < 4; ++ni)
#pragma unroll
      for (int r = 0; r < 8; ++r)
        sP[wid][r + hi8][ni * 16 + nlane] = (bf16)sacc[ni][r];

    // O += P(16x64) * V(64x128): 2 K-steps of 32 keys, 8 head-dim tiles.
    // B-fragments read from transposed V: contiguous dwords per fragment.
    const unsigned* sP32  = (const unsigned*)&sP[wid][0][0];  // 32 dwords/row
    const unsigned* sVt32 = (const unsigned*)sVt;             // 32 dwords/dim
#pragma unroll
    for (int ks2 = 0; ks2 < 2; ++ks2) {
      Frag ap;
#pragma unroll
      for (int j = 0; j < 8; ++j) {
        int kd = ks2 * 16 + gA + j;
        ap.u[j] = sP32[nlane * 32 + kd];
      }
#pragma unroll
      for (int ni = 0; ni < 8; ++ni) {
        Frag bv;
        int base = (ni * 16 + nlane) * 32 + ks2 * 16 + (gB >> 1);
#pragma unroll
        for (int j = 0; j < 8; ++j) bv.u[j] = sVt32[base + j];
        oacc[ni] = wmma_bf16(ap.v, bv.v, oacc[ni]);
      }
    }
    __syncthreads();
  }

  // normalize and store (B, S, HQ, HD) bf16
  float invl[8];
#pragma unroll
  for (int r = 0; r < 8; ++r) invl[r] = 1.0f / l[r];
#pragma unroll
  for (int ni = 0; ni < 8; ++ni)
#pragma unroll
    for (int r = 0; r < 8; ++r) {
      int qrow = qb + wid * 16 + r + hi8;
      o[(((size_t)b * S_ + qrow) * HQ_ + h) * HD_ + ni * 16 + nlane] =
          (bf16)(oacc[ni][r] * invl[r]);
    }
}

// ------------------------------------------------------------------- launcher
extern "C" void kernel_launch(void* const* d_in, const int* in_sizes, int n_in,
                              void* d_out, int out_size, void* d_ws, size_t ws_size,
                              hipStream_t stream) {
  const float* x  = (const float*)d_in[0];
  const float* wq = (const float*)d_in[1];
  const float* wk = (const float*)d_in[2];
  const float* wv = (const float*)d_in[3];
  const float* wo = (const float*)d_in[4];
  float* out = (float*)d_out;

  const size_t nX  = (size_t)B_ * S_ * DM_;        // 8M
  const size_t nWq = (size_t)DM_ * DM_;            // 4M
  const size_t nWk = (size_t)(HKV_ * HD_) * DM_;   // 2M
  const size_t M   = (size_t)B_ * S_;              // 4096

  char* w = (char*)d_ws;
  bf16* xb  = (bf16*)w;  w += nX  * 2;
  bf16* wqb = (bf16*)w;  w += nWq * 2;
  bf16* wkb = (bf16*)w;  w += nWk * 2;
  bf16* wvb = (bf16*)w;  w += nWk * 2;
  bf16* wob = (bf16*)w;  w += nWq * 2;
  bf16* qb  = (bf16*)w;  w += nX  * 2;             // (B,S,HQ,HD)
  bf16* kb  = (bf16*)w;  w += (size_t)B_ * S_ * HKV_ * HD_ * 2;
  bf16* vb  = (bf16*)w;  w += (size_t)B_ * S_ * HKV_ * HD_ * 2;
  bf16* ao  = (bf16*)w;  w += nX  * 2;             // attention out (B,S,HQ*HD)

  // 1) f32 -> bf16
  cvt_f32_bf16<<<2048, 256, 0, stream>>>(x,  xb,  (long)nX);
  cvt_f32_bf16<<<2048, 256, 0, stream>>>(wq, wqb, (long)nWq);
  cvt_f32_bf16<<<1024, 256, 0, stream>>>(wk, wkb, (long)nWk);
  cvt_f32_bf16<<<1024, 256, 0, stream>>>(wv, wvb, (long)nWk);
  cvt_f32_bf16<<<2048, 256, 0, stream>>>(wo, wob, (long)nWq);

  // 2) projections (NT GEMMs), bf16 out
  gemm_nt_bf16<true><<<dim3(DM_ / 128, M / 128), 256, 0, stream>>>(
      xb, wqb, qb, (int)M, DM_, DM_);
  gemm_nt_bf16<true><<<dim3((HKV_ * HD_) / 128, M / 128), 256, 0, stream>>>(
      xb, wkb, kb, (int)M, HKV_ * HD_, DM_);
  gemm_nt_bf16<true><<<dim3((HKV_ * HD_) / 128, M / 128), 256, 0, stream>>>(
      xb, wvb, vb, (int)M, HKV_ * HD_, DM_);

  // 3) RoPE on q and k
  {
    long nq = (long)B_ * S_ * HQ_ * (HD_ / 2);
    long nk = (long)B_ * S_ * HKV_ * (HD_ / 2);
    rope_bf16<<<(int)((nq + 255) / 256), 256, 0, stream>>>(qb, HQ_, nq);
    rope_bf16<<<(int)((nk + 255) / 256), 256, 0, stream>>>(kb, HKV_, nk);
  }

  // 4) causal flash attention (GQA)
  flash_attn_bf16<<<dim3(S_ / 64, HQ_, B_), 128, 0, stream>>>(qb, kb, vb, ao);

  // 5) output projection, f32 out
  gemm_nt_bf16<false><<<dim3(DM_ / 128, M / 128), 256, 0, stream>>>(
      ao, wob, out, (int)M, DM_, DM_);
}
